// TransformerDecoderBlock_65472481460311
// MI455X (gfx1250) — matrix operand order repaired and checked by reading, not compiled, not run on hardware
//
#include <hip/hip_runtime.h>
#include <hip/hip_bf16.h>
#include <math.h>

typedef __bf16 bf16_t;
typedef __attribute__((ext_vector_type(16))) __bf16 v16bf;
typedef __attribute__((ext_vector_type(8)))  float  v8f;

#define TPB 256

// ---------------------------------------------------------------------------
// f32 -> bf16 conversion
// ---------------------------------------------------------------------------
__global__ __launch_bounds__(TPB) void cvt_kernel(const float* __restrict__ s,
                                                  bf16_t* __restrict__ d, int n) {
  int i = blockIdx.x * TPB + threadIdx.x;
  if (i < n) d[i] = (bf16_t)s[i];
}

// ---------------------------------------------------------------------------
// CDNA5 async-copy helpers (ASYNCcnt-tracked, memory -> LDS, no VGPR bounce)
// ---------------------------------------------------------------------------
__device__ __forceinline__ void async_copy_b128(unsigned lds_addr,
                                                const void* gaddr) {
  asm volatile("global_load_async_to_lds_b128 %0, %1, off"
               :
               : "v"(lds_addr), "v"((unsigned long long)gaddr)
               : "memory");
}
__device__ __forceinline__ void wait_async_le4() {
  asm volatile("s_wait_asynccnt 0x4" ::: "memory");
}
__device__ __forceinline__ void wait_async_0() {
  asm volatile("s_wait_asynccnt 0x0" ::: "memory");
}
// Low 32 bits of a generic pointer to __shared__ == LDS byte address (ISA 10.2)
__device__ __forceinline__ unsigned lds_addr_of(const void* p) {
  return (unsigned)(unsigned long long)p;
}

// ---------------------------------------------------------------------------
// Tiled WMMA GEMM:  C[m,n] = act( alpha * sum_k A[m,k] * W[n,k] + bias[n] )
// A: bf16 [M,K] row-major (lda), W: bf16 [N,K] row-major (ldw)  ("NT" gemm)
// Outputs (any may be null): C f32 [M,N] (ldc), Cb bf16 [M,N] (ldc),
//                            CT bf16 [N,M] transposed (ldt)
// BM=128, BN=128, BK=32; 256 threads = 8 waves in a 4(M) x 2(N) grid;
// each wave computes a 32x64 tile = 2x4 v_wmma_f32_16x16x32_bf16
// (8 WMMAs per 12 ds_load_b128 -> 3/4 the LDS fragment traffic per WMMA
//  vs a 32x32 wave tile).
// Global->LDS uses global_load_async_to_lds_b128 with 2-stage double buffer:
// stage s^1 streams in (ASYNCcnt) while stage s feeds the WMMAs.
// ---------------------------------------------------------------------------
__global__ __launch_bounds__(TPB)
void gemm_nt_kernel(const bf16_t* __restrict__ A, long long sAz, int lda,
                    const bf16_t* __restrict__ W, long long sWz, int ldw,
                    const float* __restrict__ bias,
                    float* __restrict__ C, long long sCz, int ldc,
                    bf16_t* __restrict__ Cb,
                    bf16_t* __restrict__ CT, long long sCTz, int ldt,
                    int K, float alpha, int relu, int causal) {
  const int BM = 128, BN = 128, BK = 32;
  int n0 = blockIdx.x * BN;
  int m0 = blockIdx.y * BM;
  if (causal && n0 >= m0 + BM) return;  // block fully above the diagonal

  int z = blockIdx.z;
  A += (size_t)z * sAz;
  W += (size_t)z * sWz;
  if (C)  C  += (size_t)z * sCz;
  if (Cb) Cb += (size_t)z * sCz;
  if (CT) CT += (size_t)z * sCTz;

  __shared__ alignas(16) bf16_t As[2][BM * BK];  // 2 x 8 KB
  __shared__ alignas(16) bf16_t Ns[2][BN * BK];  // 2 x 8 KB

  int tid  = threadIdx.x;
  int wave = tid >> 5, lane = tid & 31;
  int half = lane >> 4, l = lane & 15;
  int wm = wave >> 1, wn = wave & 1;  // 4(M) x 2(N) wave grid

  v8f acc[2][4] = {};

  const bf16_t* Ag = A + (size_t)m0 * lda;
  const bf16_t* Wg = W + (size_t)n0 * ldw;

  // Per-thread copy geometry (16B chunks, K-major straight copy):
  // each 128x32 tile = 512 chunks -> 2 per thread.
  int r0 = tid >> 2, c0 = tid & 3;
  int r1 = (tid + TPB) >> 2;
  unsigned lA0[2], lA1[2], lN0[2], lN1[2];
#pragma unroll
  for (int s = 0; s < 2; ++s) {
    lA0[s] = lds_addr_of(&As[s][r0 * BK + c0 * 8]);
    lA1[s] = lds_addr_of(&As[s][r1 * BK + c0 * 8]);
    lN0[s] = lds_addr_of(&Ns[s][r0 * BK + c0 * 8]);
    lN1[s] = lds_addr_of(&Ns[s][r1 * BK + c0 * 8]);
  }

  auto issue = [&](int s, int kt) {
    async_copy_b128(lA0[s], &Ag[(size_t)r0 * lda + kt + c0 * 8]);
    async_copy_b128(lA1[s], &Ag[(size_t)r1 * lda + kt + c0 * 8]);
    async_copy_b128(lN0[s], &Wg[(size_t)r0 * ldw + kt + c0 * 8]);
    async_copy_b128(lN1[s], &Wg[(size_t)r1 * ldw + kt + c0 * 8]);
  };

  issue(0, 0);
  int st = 0;
  for (int kt = 0; kt < K; kt += BK, st ^= 1) {
    bool more = (kt + BK) < K;
    if (more) {
      issue(st ^ 1, kt + BK);  // stream next stage while this one computes
      wait_async_le4();        // retire previous stage (4 async ops/wave)
    } else {
      wait_async_0();
    }
    __syncthreads();  // all waves' stage-st LDS data visible

    // ---- build WMMA fragments from LDS (matches ISA VGPR layouts) ----
    union { v16bf v; uint4 q[2]; } a[2], b[4];
#pragma unroll
    for (int mi = 0; mi < 2; ++mi) {
      // A 16x32: lanes 0-15 hold K 0-7 (VGPR0-3) and K 16-23 (VGPR4-7);
      //          lanes 16-31 hold K 8-15 and K 24-31.
      const bf16_t* p = &As[st][(wm * 32 + mi * 16 + l) * BK + half * 8];
      a[mi].q[0] = *(const uint4*)(p);
      a[mi].q[1] = *(const uint4*)(p + 16);
    }
#pragma unroll
    for (int ni = 0; ni < 4; ++ni) {
      // B 32x16: lanes 0-15 hold K 0-7 and K 16-23; lanes 16-31 hold K 8-15 and K 24-31.
      const bf16_t* p = &Ns[st][(wn * 64 + ni * 16 + l) * BK + half * 8];
      b[ni].q[0] = *(const uint4*)(p);
      b[ni].q[1] = *(const uint4*)(p + 16);
    }
#pragma unroll
    for (int mi = 0; mi < 2; ++mi)
#pragma unroll
      for (int ni = 0; ni < 4; ++ni)
        acc[mi][ni] = __builtin_amdgcn_wmma_f32_16x16x32_bf16(
            false, a[mi].v, false, b[ni].v, (short)0, acc[mi][ni], false, false);

    __syncthreads();  // protect buffer st before it is refilled next iter
  }

  // ---- epilogue: alpha, bias, relu; f32 / bf16 / bf16-transposed stores ----
#pragma unroll
  for (int ni = 0; ni < 4; ++ni) {
    int col = n0 + wn * 64 + ni * 16 + l;
    float bv = bias ? bias[col] : 0.0f;
#pragma unroll
    for (int mi = 0; mi < 2; ++mi) {
      int rowBase = m0 + wm * 32 + mi * 16 + half * 8;
      union { bf16_t h[8]; uint4 q; } tcol;
#pragma unroll
      for (int r = 0; r < 8; ++r) {
        float v = acc[mi][ni][r] * alpha + bv;
        if (relu) v = fmaxf(v, 0.0f);
        if (C)  C[(size_t)(rowBase + r) * ldc + col] = v;
        if (Cb) Cb[(size_t)(rowBase + r) * ldc + col] = (bf16_t)v;
        tcol.h[r] = (bf16_t)v;
      }
      if (CT) *(uint4*)(&CT[(size_t)col * ldt + rowBase]) = tcol.q;
    }
  }
}

// ---------------------------------------------------------------------------
// Row softmax over S=2048 columns, optional causal mask; f32 in, bf16 out.
// One 256-thread block per row; masked entries produce exact bf16 zeros.
// ---------------------------------------------------------------------------
__device__ inline float waveMax(float v) {
#pragma unroll
  for (int m = 16; m; m >>= 1) v = fmaxf(v, __shfl_xor(v, m, 32));
  return v;
}
__device__ inline float waveSum(float v) {
#pragma unroll
  for (int m = 16; m; m >>= 1) v += __shfl_xor(v, m, 32);
  return v;
}

__global__ __launch_bounds__(TPB)
void softmax_kernel(const float* __restrict__ Sc, bf16_t* __restrict__ P,
                    int causal) {
  const int S = 2048, PER = S / TPB;  // 8
  int row = blockIdx.x;               // 0 .. B*S-1 over contiguous [B][S][S]
  int q = row & (S - 1);
  size_t base = (size_t)row * S;
  int tid = threadIdx.x, lane = tid & 31, wave = tid >> 5;
  __shared__ float red[8];

  float ev[PER];
  float lmax = -3.0e38f;
#pragma unroll
  for (int i = 0; i < PER; ++i) {
    int j = tid + i * TPB;
    float v = (causal && j > q) ? -3.0e38f : Sc[base + j];
    ev[i] = v;
    lmax = fmaxf(lmax, v);
  }
  lmax = waveMax(lmax);
  if (lane == 0) red[wave] = lmax;
  __syncthreads();
  float gmax = red[0];
#pragma unroll
  for (int w = 1; w < 8; ++w) gmax = fmaxf(gmax, red[w]);
  __syncthreads();

  float lsum = 0.f;
#pragma unroll
  for (int i = 0; i < PER; ++i) {
    ev[i] = __expf(ev[i] - gmax);  // masked -> exp(-huge) == 0
    lsum += ev[i];
  }
  lsum = waveSum(lsum);
  if (lane == 0) red[wave] = lsum;
  __syncthreads();
  float gsum = 0.f;
#pragma unroll
  for (int w = 0; w < 8; ++w) gsum += red[w];
  float inv = 1.0f / gsum;
#pragma unroll
  for (int i = 0; i < PER; ++i)
    P[base + tid + i * TPB] = (bf16_t)(ev[i] * inv);
}

// ---------------------------------------------------------------------------
// Residual add + LayerNorm over E=768; f32 out + optional bf16 out.
// One 256-thread block per token row.
// ---------------------------------------------------------------------------
__global__ __launch_bounds__(TPB)
void resln_kernel(const float* __restrict__ X, const float* __restrict__ H,
                  const float* __restrict__ g, const float* __restrict__ be,
                  float* __restrict__ O, bf16_t* __restrict__ Ob) {
  const int E = 768, PER = E / TPB;  // 3
  size_t base = (size_t)blockIdx.x * E;
  int tid = threadIdx.x, lane = tid & 31, wave = tid >> 5;
  __shared__ float red[8];

  float t[PER];
  float lsum = 0.f;
#pragma unroll
  for (int i = 0; i < PER; ++i) {
    int c = tid + i * TPB;
    t[i] = X[base + c] + H[base + c];
    lsum += t[i];
  }
  lsum = waveSum(lsum);
  if (lane == 0) red[wave] = lsum;
  __syncthreads();
  float mu = 0.f;
#pragma unroll
  for (int w = 0; w < 8; ++w) mu += red[w];
  mu *= (1.0f / E);
  __syncthreads();

  float lv = 0.f;
#pragma unroll
  for (int i = 0; i < PER; ++i) {
    float d = t[i] - mu;
    lv += d * d;
  }
  lv = waveSum(lv);
  if (lane == 0) red[wave] = lv;
  __syncthreads();
  float var = 0.f;
#pragma unroll
  for (int w = 0; w < 8; ++w) var += red[w];
  float rs = rsqrtf(var * (1.0f / E) + 1e-5f);

#pragma unroll
  for (int i = 0; i < PER; ++i) {
    int c = tid + i * TPB;
    float y = (t[i] - mu) * rs * g[c] + be[c];
    O[base + c] = y;
    if (Ob) Ob[base + c] = (bf16_t)y;
  }
}

// ---------------------------------------------------------------------------
// Host-side orchestration
// ---------------------------------------------------------------------------
extern "C" void kernel_launch(void* const* d_in, const int* in_sizes, int n_in,
                              void* d_out, int out_size, void* d_ws,
                              size_t ws_size, hipStream_t stream) {
  (void)in_sizes; (void)n_in; (void)out_size; (void)ws_size;
  const int Bq = 4, S = 2048, E = 768, DFF = 3072, M = Bq * S;  // M = 8192
  const float scal = 1.0f / sqrtf((float)E);

  const float* x    = (const float*)d_in[0];
  const float* kv   = (const float*)d_in[1];
  const float* wq_w = (const float*)d_in[2];  const float* wq_b = (const float*)d_in[3];
  const float* wk_w = (const float*)d_in[4];  const float* wk_b = (const float*)d_in[5];
  const float* wv_w = (const float*)d_in[6];  const float* wv_b = (const float*)d_in[7];
  const float* ln1_g = (const float*)d_in[8]; const float* ln1_b = (const float*)d_in[9];
  const float* wq2_w = (const float*)d_in[10]; const float* wq2_b = (const float*)d_in[11];
  const float* wk2_w = (const float*)d_in[12]; const float* wk2_b = (const float*)d_in[13];
  const float* wv2_w = (const float*)d_in[14]; const float* wv2_b = (const float*)d_in[15];
  const float* ln2_g = (const float*)d_in[16]; const float* ln2_b = (const float*)d_in[17];
  const float* w1 = (const float*)d_in[18]; const float* b1 = (const float*)d_in[19];
  const float* w2 = (const float*)d_in[20]; const float* b2 = (const float*)d_in[21];
  const float* ln3_g = (const float*)d_in[22]; const float* ln3_b = (const float*)d_in[23];
  float* out = (float*)d_out;

  // ---- workspace carve-up (fixed offsets, 256B aligned) ----
  char* ws = (char*)d_ws;
  size_t off = 0;
  auto alloc = [&](size_t bytes) -> char* {
    char* p = ws + off;
    off = (off + bytes + 255) & ~(size_t)255;
    return p;
  };
  bf16_t* wqb  = (bf16_t*)alloc((size_t)E * E * 2);
  bf16_t* wkb  = (bf16_t*)alloc((size_t)E * E * 2);
  bf16_t* wvb  = (bf16_t*)alloc((size_t)E * E * 2);
  bf16_t* wq2b = (bf16_t*)alloc((size_t)E * E * 2);
  bf16_t* wk2b = (bf16_t*)alloc((size_t)E * E * 2);
  bf16_t* wv2b = (bf16_t*)alloc((size_t)E * E * 2);
  bf16_t* w1b  = (bf16_t*)alloc((size_t)DFF * E * 2);
  bf16_t* w2b  = (bf16_t*)alloc((size_t)E * DFF * 2);
  bf16_t* xb   = (bf16_t*)alloc((size_t)M * E * 2);
  bf16_t* kvb  = (bf16_t*)alloc((size_t)M * E * 2);
  bf16_t* qb   = (bf16_t*)alloc((size_t)M * E * 2);
  bf16_t* kb   = (bf16_t*)alloc((size_t)M * E * 2);
  bf16_t* vTb  = (bf16_t*)alloc((size_t)E * M * 2);   // [E][B*S] transposed
  float*  sc   = (float*) alloc((size_t)Bq * S * S * 4);
  bf16_t* Pb   = (bf16_t*)alloc((size_t)Bq * S * S * 2);
  float*  h    = (float*) alloc((size_t)M * E * 4);
  float*  xs1  = (float*) alloc((size_t)M * E * 4);
  float*  xs2  = (float*) alloc((size_t)M * E * 4);
  bf16_t* xsb  = (bf16_t*)alloc((size_t)M * E * 2);
  bf16_t* hm   = (bf16_t*)alloc((size_t)M * DFF * 2);

  auto cvt = [&](const float* s, bf16_t* d, int n) {
    cvt_kernel<<<(n + TPB - 1) / TPB, TPB, 0, stream>>>(s, d, n);
  };
  auto gemm = [&](const bf16_t* A, long long sAz, int lda,
                  const bf16_t* W, long long sWz, int ldw, const float* bias,
                  float* C, long long sCz, int ldc, bf16_t* Cb, bf16_t* CT,
                  long long sCTz, int ldt, int Mm, int Nn, int K, int Z,
                  float alpha, int relu, int causal) {
    dim3 g(Nn / 128, Mm / 128, Z);
    gemm_nt_kernel<<<g, TPB, 0, stream>>>(A, sAz, lda, W, sWz, ldw, bias, C,
                                          sCz, ldc, Cb, CT, sCTz, ldt, K,
                                          alpha, relu, causal);
  };

  // ---- 0. convert inputs & weights to bf16 ----
  cvt(x, xb, M * E);
  cvt(kv, kvb, M * E);
  cvt(wq_w, wqb, E * E);   cvt(wk_w, wkb, E * E);   cvt(wv_w, wvb, E * E);
  cvt(wq2_w, wq2b, E * E); cvt(wk2_w, wk2b, E * E); cvt(wv2_w, wv2b, E * E);
  cvt(w1, w1b, DFF * E);   cvt(w2, w2b, E * DFF);

  // ---- 1. causal self-attention ----
  gemm(xb, 0, E, wqb, 0, E, wq_b, nullptr, 0, E, qb, nullptr, 0, 0,
       M, E, E, 1, 1.0f, 0, 0);                              // Q (bf16)
  gemm(xb, 0, E, wkb, 0, E, wk_b, nullptr, 0, E, kb, nullptr, 0, 0,
       M, E, E, 1, 1.0f, 0, 0);                              // K (bf16)
  gemm(xb, 0, E, wvb, 0, E, wv_b, nullptr, 0, E, nullptr, vTb, 0, M,
       M, E, E, 1, 1.0f, 0, 0);                              // V^T (bf16 [E][M])
  gemm(qb, (long long)S * E, E, kb, (long long)S * E, E, nullptr,
       sc, (long long)S * S, S, nullptr, nullptr, 0, 0,
       S, S, E, Bq, scal, 0, /*causal=*/1);                  // scores
  softmax_kernel<<<M, TPB, 0, stream>>>(sc, Pb, 1);          // P (bf16)
  gemm(Pb, (long long)S * S, S, vTb, (long long)S, M, nullptr,
       h, (long long)S * E, E, nullptr, nullptr, 0, 0,
       S, E, S, Bq, 1.0f, 0, 0);                             // H1 = P @ V
  resln_kernel<<<M, TPB, 0, stream>>>(x, h, ln1_g, ln1_b, xs1, xsb);

  // ---- 2. cross-attention over kv ----
  gemm(xsb, 0, E, wq2b, 0, E, wq2_b, nullptr, 0, E, qb, nullptr, 0, 0,
       M, E, E, 1, 1.0f, 0, 0);                              // Q2
  gemm(kvb, 0, E, wk2b, 0, E, wk2_b, nullptr, 0, E, kb, nullptr, 0, 0,
       M, E, E, 1, 1.0f, 0, 0);                              // K2
  gemm(kvb, 0, E, wv2b, 0, E, wv2_b, nullptr, 0, E, nullptr, vTb, 0, M,
       M, E, E, 1, 1.0f, 0, 0);                              // V2^T
  gemm(qb, (long long)S * E, E, kb, (long long)S * E, E, nullptr,
       sc, (long long)S * S, S, nullptr, nullptr, 0, 0,
       S, S, E, Bq, scal, 0, 0);                             // scores2 (no mask)
  softmax_kernel<<<M, TPB, 0, stream>>>(sc, Pb, 0);
  gemm(Pb, (long long)S * S, S, vTb, (long long)S, M, nullptr,
       h, (long long)S * E, E, nullptr, nullptr, 0, 0,
       S, E, S, Bq, 1.0f, 0, 0);                             // H2
  resln_kernel<<<M, TPB, 0, stream>>>(xs1, h, ln2_g, ln2_b, xs2, xsb);

  // ---- 3. MLP: relu(x @ w1.T + b1) -> relu(h @ w2.T + b2) ----
  gemm(xsb, 0, E, w1b, 0, E, b1, nullptr, 0, DFF, hm, nullptr, 0, 0,
       M, DFF, E, 1, 1.0f, /*relu=*/1, 0);                   // hidden (bf16)
  gemm(hm, 0, DFF, w2b, 0, DFF, b2, h, 0, E, nullptr, nullptr, 0, 0,
       M, E, DFF, 1, 1.0f, /*relu=*/1, 0);                   // mlp (f32)
  resln_kernel<<<M, TPB, 0, stream>>>(xs2, h, ln3_g, ln3_b, out, nullptr);
}
